// JointAttnBlock_91087666413727
// MI455X (gfx1250) — matrix operand order repaired and checked by reading, not compiled, not run on hardware
//
#include <hip/hip_runtime.h>
#include <hip/hip_bf16.h>

typedef __attribute__((ext_vector_type(16))) __bf16 v16bf;
typedef __attribute__((ext_vector_type(8)))  float  v8f;
typedef __attribute__((ext_vector_type(4)))  unsigned int v4u;

#define B_   32
#define CIN  512
#define T_   1024
#define CHID 832
#define J_   52
#define NGRP 32

// Flip to 0 if the assembler rejects global_load_async_to_lds_b128.
#define USE_ASYNC_LDS 1

// ---------------- workspace byte offsets ----------------
#define OFF_STATS 0u
#define OFF_WQ    8192u                       // 3x (832*512) bf16, consecutive
#define WSZ       (832u*512u*2u)              // 851968
#define OFF_WP    (OFF_WQ + 3u*WSZ)           // 2564096
#define OFF_HT    3416064u                    // (B,T,CIN) bf16   33.5MB
#define OFF_Q     36970496u                   // (B,T,CHID) bf16  54.5MB each, Q/K/V consecutive
#define QSZ       ((size_t)B_*T_*CHID*2u)
#define OFF_HO    (OFF_Q + 3u*QSZ)            // 200548352

// ---------------- async copy helpers (CDNA5 ASYNCcnt path) ----------------
__device__ __forceinline__ void async_ld_b128(unsigned lds_off, const void* gaddr) {
#if USE_ASYNC_LDS
  asm volatile("global_load_async_to_lds_b128 %0, %1, off"
               :: "v"(lds_off), "v"(gaddr) : "memory");
#endif
}
__device__ __forceinline__ void wait_asynccnt0() {
#if USE_ASYNC_LDS
  asm volatile("s_wait_asynccnt 0x0" ::: "memory");
#endif
}
__device__ __forceinline__ unsigned lds_offset_of(const void* p) {
  // generic pointer to LDS: low 32 bits are the LDS byte offset
  return (unsigned)(unsigned long long)p;
}

// ---------------- f32 -> bf16 conversion ----------------
__global__ __launch_bounds__(256) void k_f2bf(const float* __restrict__ in,
                                              __hip_bfloat16* __restrict__ out, int n) {
  int i = blockIdx.x * 256 + threadIdx.x;
  if (i < n) out[i] = __float2bfloat16(in[i]);
}

// ---------------- groupnorm stats: one block per (b,g) ----------------
__global__ __launch_bounds__(256) void k_gnstats(const float* __restrict__ x,
                                                 float* __restrict__ stats) {
  int bg = blockIdx.x;                       // group data is contiguous 16384 floats
  const float4* base = (const float4*)(x + (size_t)bg * 16384);
  float s = 0.f, ss = 0.f;
  for (int i = threadIdx.x; i < 4096; i += 256) {
    float4 v = base[i];
    s  += v.x + v.y + v.z + v.w;
    ss += v.x*v.x + v.y*v.y + v.z*v.z + v.w*v.w;
  }
  for (int off = 16; off > 0; off >>= 1) {
    s  += __shfl_down(s,  off, 32);
    ss += __shfl_down(ss, off, 32);
  }
  __shared__ float ps[8], pss[8];
  int w = threadIdx.x >> 5;
  if ((threadIdx.x & 31) == 0) { ps[w] = s; pss[w] = ss; }
  __syncthreads();
  if (threadIdx.x == 0) {
    float ts = 0.f, tss = 0.f;
    for (int i = 0; i < 8; ++i) { ts += ps[i]; tss += pss[i]; }
    float mu  = ts * (1.f / 16384.f);
    float var = tss * (1.f / 16384.f) - mu * mu;
    stats[2*bg]   = mu;
    stats[2*bg+1] = rsqrtf(var + 1e-6f);
  }
}

// ---------------- GN apply + transpose: x(b,c,t) -> h_t(b,t,c) bf16 ----------------
__global__ __launch_bounds__(256) void k_gnapply(const float* __restrict__ x,
                                                 const float* __restrict__ stats,
                                                 const float* __restrict__ sc,
                                                 const float* __restrict__ bi,
                                                 __hip_bfloat16* __restrict__ Ht) {
  int c0 = blockIdx.x * 32, t0 = blockIdx.y * 32, b = blockIdx.z;
  __shared__ float tile[32][33];
  #pragma unroll
  for (int i = 0; i < 4; ++i) {
    int c = c0 + threadIdx.y + i * 8;
    int t = t0 + threadIdx.x;
    float mu = stats[2*(b*NGRP + (c >> 4))];
    float rs = stats[2*(b*NGRP + (c >> 4)) + 1];
    float v  = x[((size_t)b*CIN + c)*T_ + t];
    tile[threadIdx.y + i*8][threadIdx.x] = (v - mu) * rs * sc[c] + bi[c];
  }
  __syncthreads();
  #pragma unroll
  for (int i = 0; i < 4; ++i) {
    int t = t0 + threadIdx.y + i * 8;
    int c = c0 + threadIdx.x;
    Ht[((size_t)b*T_ + t)*CIN + c] = __float2bfloat16(tile[threadIdx.x][threadIdx.y + i*8]);
  }
}

// ---------------- fragment helpers (ISA 7.12.2 layouts) ----------------
union FragBF { v16bf f; v4u u[2]; };
union Pack8  { __hip_bfloat16 h[8]; v4u u; };

// A tile in LDS: [m][k] 64x32 bf16 (64B rows). B tile: [n][k] 128x32 bf16.
__device__ __forceinline__ void load_afrag(FragBF& a, const char* ldsA, int mrow, int lane) {
  int m = lane & 15, hi = lane >> 4;
  const char* p = ldsA + (mrow + m) * 64 + hi * 16;
  a.u[0] = *(const v4u*)(p);
  a.u[1] = *(const v4u*)(p + 32);
}
__device__ __forceinline__ void load_bfrag(FragBF& bfr, const char* ldsB, int ncol, int lane) {
  int n = lane & 15, hi = lane >> 4;
  const char* p = ldsB + (ncol + n) * 64 + hi * 16;
  bfr.u[0] = *(const v4u*)(p);
  bfr.u[1] = *(const v4u*)(p + 32);
}

// Issue loads of one K-step tile (A: 64x32, B: 128x32) into a double-buffer slot.
// STRIDE = row stride in bytes for both A and B (equal for our shapes).
template<int STRIDE>
__device__ __forceinline__ void fill_tiles(char* bufA, char* bufB,
                                           const char* gA, const char* gB,
                                           int k0, int tid) {
  #pragma unroll
  for (int it = 0; it < 2; ++it) {                       // A: 256 x 16B chunks
    int idx = tid + it * 128;
    int row = idx >> 2, c4 = idx & 3;
    size_t go = (size_t)row * STRIDE + (size_t)k0 * 2 + c4 * 16;
#if USE_ASYNC_LDS
    async_ld_b128(lds_offset_of(bufA + row*64 + c4*16), gA + go);
#else
    *(v4u*)(bufA + row*64 + c4*16) = *(const v4u*)(gA + go);
#endif
  }
  #pragma unroll
  for (int it = 0; it < 4; ++it) {                       // B: 512 x 16B chunks
    int idx = tid + it * 128;
    int row = idx >> 2, c4 = idx & 3;
    size_t go = (size_t)row * STRIDE + (size_t)k0 * 2 + c4 * 16;
#if USE_ASYNC_LDS
    async_ld_b128(lds_offset_of(bufB + row*64 + c4*16), gB + go);
#else
    *(v4u*)(bufB + row*64 + c4*16) = *(const v4u*)(gB + go);
#endif
  }
}

// One K-step of math: 2x4 accumulator tiles per wave (32M x 64N).
__device__ __forceinline__ void compute_step(const char* bufA, const char* bufB,
                                             int wm, int wn, int lane, v8f acc[2][4]) {
  FragBF af[2], bf[4];
  #pragma unroll
  for (int r = 0; r < 2; ++r) load_afrag(af[r], bufA, wm*32 + r*16, lane);
  #pragma unroll
  for (int s = 0; s < 4; ++s) load_bfrag(bf[s], bufB, wn*64 + s*16, lane);
  #pragma unroll
  for (int r = 0; r < 2; ++r)
    #pragma unroll
    for (int s = 0; s < 4; ++s)
      acc[r][s] = __builtin_amdgcn_wmma_f32_16x16x32_bf16(
          false, af[r].f, false, bf[s].f, (short)0, acc[r][s], false, false);
}

#define BUFBYTES 12288                                   // A 4KB + B 8KB per slot

// ---------------- fused QKV GEMM: out(b,t,832) = W(832x512) @ h_t^T ----------------
__global__ __launch_bounds__(128) void k_gemm_qkv(const __hip_bfloat16* __restrict__ Wall,
                                                  const __hip_bfloat16* __restrict__ Ht,
                                                  const float* __restrict__ bq,
                                                  const float* __restrict__ bk,
                                                  const float* __restrict__ bv,
                                                  __hip_bfloat16* __restrict__ Qall) {
  const int KD = CIN;                                    // 512, 16 K-steps
  int mat = blockIdx.x / 13;
  int mt  = blockIdx.x % 13;
  int mBase = mt * 64;
  int tBase = blockIdx.y * 128;
  int b = blockIdx.z;
  const char* gA = (const char*)(Wall + (size_t)mat * (CHID * CIN)) + (size_t)mBase * KD * 2;
  const char* gB = (const char*)(Ht + ((size_t)b * T_ + tBase) * CIN);
  const float* bias = (mat == 0) ? bq : (mat == 1 ? bk : bv);
  __hip_bfloat16* Out = Qall + (size_t)mat * ((size_t)B_ * T_ * CHID);

  __shared__ __align__(16) char smem[2 * BUFBYTES];      // double buffered A+B
  int tid = threadIdx.x, lane = tid & 31, w = tid >> 5;
  int wm = w >> 1, wn = w & 1;                           // 2x2 waves, 32M x 64N each
  v8f acc[2][4] = {};

  fill_tiles<CIN*2>(smem, smem + 4096, gA, gB, 0, tid);
  wait_asynccnt0();
  __syncthreads();
  int cur = 0;
  for (int i = 0; i < KD / 32; ++i) {
    char* bc = smem + cur * BUFBYTES;
    if (i + 1 < KD / 32) {
      char* bn = smem + (cur ^ 1) * BUFBYTES;
      fill_tiles<CIN*2>(bn, bn + 4096, gA, gB, (i + 1) * 32, tid);
    }
    compute_step(bc, bc + 4096, wm, wn, lane, acc);
    wait_asynccnt0();
    __syncthreads();
    cur ^= 1;
  }

  // D layout: lanes0-15 N=lane M=r ; lanes16-31 N=lane-16 M=r+8 (8 consecutive M per lane)
  int n = lane & 15, mo = (lane >> 4) * 8;
  #pragma unroll
  for (int r = 0; r < 2; ++r)
    #pragma unroll
    for (int s = 0; s < 4; ++s) {
      int ch = mBase + wm*32 + r*16 + mo;
      int t  = tBase + wn*64 + s*16 + n;
      float4 b0 = *(const float4*)(bias + ch);
      float4 b1 = *(const float4*)(bias + ch + 4);
      Pack8 p;
      p.h[0] = __float2bfloat16(acc[r][s][0] + b0.x);
      p.h[1] = __float2bfloat16(acc[r][s][1] + b0.y);
      p.h[2] = __float2bfloat16(acc[r][s][2] + b0.z);
      p.h[3] = __float2bfloat16(acc[r][s][3] + b0.w);
      p.h[4] = __float2bfloat16(acc[r][s][4] + b1.x);
      p.h[5] = __float2bfloat16(acc[r][s][5] + b1.y);
      p.h[6] = __float2bfloat16(acc[r][s][6] + b1.z);
      p.h[7] = __float2bfloat16(acc[r][s][7] + b1.w);
      *(v4u*)(Out + ((size_t)b * T_ + t) * CHID + ch) = p.u;
    }
}

// ---------------- per-(b,t) joint attention ----------------
__global__ __launch_bounds__(128) void k_attn(const __hip_bfloat16* __restrict__ Qt,
                                              const __hip_bfloat16* __restrict__ Kt,
                                              const __hip_bfloat16* __restrict__ Vt,
                                              __hip_bfloat16* __restrict__ HOt) {
  int t = blockIdx.x, b = blockIdx.y;
  size_t row = ((size_t)b * T_ + t) * CHID;
  __shared__ float qs[CHID], ks[CHID], vs[CHID], sm[J_ * J_];
  for (int i = threadIdx.x; i < CHID; i += 128) {
    qs[i] = __bfloat162float(Qt[row + i]);
    ks[i] = __bfloat162float(Kt[row + i]);
    vs[i] = __bfloat162float(Vt[row + i]);
  }
  __syncthreads();
  for (int idx = threadIdx.x; idx < J_ * J_; idx += 128) {
    int j = idx / J_, i2 = idx % J_;
    float a = 0.f;
    #pragma unroll
    for (int c = 0; c < 16; ++c) a += qs[j*16 + c] * ks[i2*16 + c];
    sm[idx] = a * 0.25f;                                 // c^-0.5 = 1/4
  }
  __syncthreads();
  if (threadIdx.x < J_) {
    int j = threadIdx.x;
    float mx = -1e30f;
    for (int i2 = 0; i2 < J_; ++i2) mx = fmaxf(mx, sm[j*J_ + i2]);
    float su = 0.f;
    for (int i2 = 0; i2 < J_; ++i2) { float e = __expf(sm[j*J_ + i2] - mx); sm[j*J_ + i2] = e; su += e; }
    float inv = 1.f / su;
    for (int i2 = 0; i2 < J_; ++i2) sm[j*J_ + i2] *= inv;
  }
  __syncthreads();
  // out channel (i*16+c) = sum_j v[j*16+c] * w[j][i]
  for (int idx = threadIdx.x; idx < CHID; idx += 128) {
    int i2 = idx >> 4, c = idx & 15;
    float a = 0.f;
    for (int j = 0; j < J_; ++j) a += vs[j*16 + c] * sm[j*J_ + i2];
    HOt[row + idx] = __float2bfloat16(a);
  }
}

// ---------------- projection GEMM + bias + residual: out(b,512,t) ----------------
__global__ __launch_bounds__(128) void k_gemm_proj(const __hip_bfloat16* __restrict__ Wp,
                                                   const __hip_bfloat16* __restrict__ HOt,
                                                   const float* __restrict__ bp,
                                                   const float* __restrict__ x,
                                                   float* __restrict__ out) {
  const int KD = CHID;                                   // 832, 26 K-steps
  int mBase = blockIdx.x * 64;
  int tBase = blockIdx.y * 128;
  int b = blockIdx.z;
  const char* gA = (const char*)Wp + (size_t)mBase * KD * 2;
  const char* gB = (const char*)(HOt + ((size_t)b * T_ + tBase) * CHID);

  // loop: two 12KB A+B buffers; epilogue: 64x129 f32 out-tile (33024B)
  __shared__ __align__(16) char smem[64 * 129 * 4];
  int tid = threadIdx.x, lane = tid & 31, w = tid >> 5;
  int wm = w >> 1, wn = w & 1;
  v8f acc[2][4] = {};

  fill_tiles<CHID*2>(smem, smem + 4096, gA, gB, 0, tid);
  wait_asynccnt0();
  __syncthreads();
  int cur = 0;
  for (int i = 0; i < KD / 32; ++i) {
    char* bc = smem + cur * BUFBYTES;
    if (i + 1 < KD / 32) {
      char* bn = smem + (cur ^ 1) * BUFBYTES;
      fill_tiles<CHID*2>(bn, bn + 4096, gA, gB, (i + 1) * 32, tid);
    }
    compute_step(bc, bc + 4096, wm, wn, lane, acc);
    wait_asynccnt0();
    __syncthreads();
    cur ^= 1;
  }

  float (*ot)[129] = (float(*)[129])smem;
  int n = lane & 15, mo = (lane >> 4) * 8;
  #pragma unroll
  for (int r = 0; r < 2; ++r)
    #pragma unroll
    for (int s = 0; s < 4; ++s) {
      int m0 = wm*32 + r*16 + mo;
      int n0 = wn*64 + s*16 + n;
      #pragma unroll
      for (int i = 0; i < 8; ++i) ot[m0 + i][n0] = acc[r][s][i];
    }
  __syncthreads();
  for (int idx = tid; idx < 64 * 128; idx += 128) {
    int row = idx >> 7, col = idx & 127;
    int m = mBase + row, t = tBase + col;
    size_t g = ((size_t)b * CIN + m) * T_ + t;
    out[g] = ot[row][col] + bp[m] + x[g];
  }
}

// ---------------- host launch ----------------
extern "C" void kernel_launch(void* const* d_in, const int* in_sizes, int n_in,
                              void* d_out, int out_size, void* d_ws, size_t ws_size,
                              hipStream_t stream) {
  const float* x   = (const float*)d_in[0];
  const float* gns = (const float*)d_in[1];
  const float* gnb = (const float*)d_in[2];
  const float* wq  = (const float*)d_in[3];
  const float* bq  = (const float*)d_in[4];
  const float* wk  = (const float*)d_in[5];
  const float* bk  = (const float*)d_in[6];
  const float* wv  = (const float*)d_in[7];
  const float* bv  = (const float*)d_in[8];
  const float* wp  = (const float*)d_in[9];
  const float* bp  = (const float*)d_in[10];

  char* ws = (char*)d_ws;
  float*          stats = (float*)(ws + OFF_STATS);
  __hip_bfloat16* wqb   = (__hip_bfloat16*)(ws + OFF_WQ);       // wq,wk,wv consecutive
  __hip_bfloat16* wpb   = (__hip_bfloat16*)(ws + OFF_WP);
  __hip_bfloat16* Ht    = (__hip_bfloat16*)(ws + OFF_HT);
  __hip_bfloat16* Qt    = (__hip_bfloat16*)(ws + OFF_Q);        // Q,K,V consecutive
  __hip_bfloat16* Kt    = Qt + (size_t)B_ * T_ * CHID;
  __hip_bfloat16* Vt    = Kt + (size_t)B_ * T_ * CHID;
  __hip_bfloat16* HOt   = (__hip_bfloat16*)(ws + OFF_HO);

  const int WN = CHID * CIN;                                    // 425984
  k_f2bf<<<(WN + 255) / 256, 256, 0, stream>>>(wq, wqb,            WN);
  k_f2bf<<<(WN + 255) / 256, 256, 0, stream>>>(wk, wqb + WN,       WN);
  k_f2bf<<<(WN + 255) / 256, 256, 0, stream>>>(wv, wqb + 2 * WN,   WN);
  k_f2bf<<<(WN + 255) / 256, 256, 0, stream>>>(wp, wpb,            WN);

  k_gnstats<<<B_ * NGRP, 256, 0, stream>>>(x, stats);
  k_gnapply<<<dim3(CIN/32, T_/32, B_), dim3(32, 8), 0, stream>>>(x, stats, gns, gnb, Ht);

  k_gemm_qkv<<<dim3(13 * 3, T_/128, B_), 128, 0, stream>>>(wqb, Ht, bq, bk, bv, Qt);
  k_attn<<<dim3(T_, B_), 128, 0, stream>>>(Qt, Kt, Vt, HOt);
  k_gemm_proj<<<dim3(CIN/64, T_/128, B_), 128, 0, stream>>>(wpb, HOt, bp, x, (float*)d_out);
}
